// Model_4690104287834
// MI455X (gfx1250) — matrix operand order repaired and checked by reading, not compiled, not run on hardware
//
#include <hip/hip_runtime.h>
#include <hip/hip_bf16.h>
#include <stdint.h>

#define CODE  2048
#define ARRAY 512
#define LENS  8192
#define LDA   40   // LDS row stride in bf16 elements (80B: 16B-aligned, bank-conflict-friendly)

typedef __attribute__((ext_vector_type(16))) __bf16 bf16x16;
typedef __attribute__((ext_vector_type(8)))  __bf16 bf16x8;
typedef __attribute__((ext_vector_type(8)))  float  f32x8;

static __device__ __forceinline__ f32x8 wmma_acc(bf16x16 a, bf16x16 b, f32x8 c) {
  // (neg_a, A, neg_b, B, c_mod, C, reuse_a, reuse_b) — neg_a/neg_b must be 0 for bf16
  return __builtin_amdgcn_wmma_f32_16x16x32_bf16(false, a, false, b, (short)0, c, false, false);
}

// A-fragment (16x32 bf16): lane holds row m, two 8-elem k-chunks at lh*8 and lh*8+16 (ISA layout)
static __device__ __forceinline__ bf16x16 ld_frag_a(const __bf16* s, int row, int lh) {
  const bf16x8 lo = *(const bf16x8*)(s + row * LDA + lh * 8);
  const bf16x8 hi = *(const bf16x8*)(s + row * LDA + lh * 8 + 16);
  bf16x16 r;
  #pragma unroll
  for (int i = 0; i < 8; ++i) { r[i] = lo[i]; r[i + 8] = hi[i]; }
  return r;
}

// B-fragment (32x16 bf16) from N-major LDS tile: lane holds col n, two 8-elem k-chunks at lh*8 and lh*8+16 (ISA layout)
static __device__ __forceinline__ bf16x16 ld_frag_b(const __bf16* s, int row, int lh) {
  const bf16x8 lo = *(const bf16x8*)(s + row * LDA + lh * 8);
  const bf16x8 hi = *(const bf16x8*)(s + row * LDA + lh * 8 + 16);
  bf16x16 r;
  #pragma unroll
  for (int i = 0; i < 8; ++i) { r[i] = lo[i]; r[i + 8] = hi[i]; }
  return r;
}

static __device__ __forceinline__ void st_row32(__bf16* p, bf16x16 v) {
  bf16x8 lo, hi;
  #pragma unroll
  for (int i = 0; i < 8; ++i) { lo[i] = v[i]; hi[i] = v[i + 8]; }
  *(bf16x8*)p = lo;
  *(bf16x8*)(p + 8) = hi;
}

// ---------------- prep: normalize w -> bf16, normalize theta -> f32x2 ----------------
__global__ __launch_bounds__(256) void k_prep_w(const float* __restrict__ wr,
                                                const float* __restrict__ wi,
                                                __bf16* __restrict__ onr,
                                                __bf16* __restrict__ oni) {
  const int i = blockIdx.x * 256 + threadIdx.x;
  const float re = wr[i], im = wi[i];
  const float inv = rsqrtf(re * re + im * im);
  onr[i] = (__bf16)(re * inv);
  oni[i] = (__bf16)(im * inv);
}

__global__ __launch_bounds__(256) void k_prep_theta(const float* __restrict__ tr,
                                                    const float* __restrict__ ti,
                                                    float2* __restrict__ out) {
  const int i = blockIdx.x * 256 + threadIdx.x;
  const float re = tr[i], im = ti[i];
  const float inv = rsqrtf(re * re + im * im);
  out[i] = make_float2(re * inv, im * inv);
}

// ---------------- GEMM1: lens = (w_n @ G) * diag(theta_n), bf16 output ----------------
// block tile 128(M) x 64(N), 8 waves (4x2), 32x32 per wave, K-chunk 32, double-buffered LDS
__global__ __launch_bounds__(256) void k_gemm1(
    const __bf16* __restrict__ Wr, const __bf16* __restrict__ Wi,
    const float*  __restrict__ Gr, const float*  __restrict__ Gi,
    const float2* __restrict__ Th,
    __bf16* __restrict__ Lr, __bf16* __restrict__ Li) {
  __shared__ __bf16 sW[2][2][128 * LDA];
  __shared__ __bf16 sB[2][2][64 * LDA];

  const int tid = threadIdx.x;
  const int lane = tid & 31, wv = tid >> 5;
  const int ln = lane & 15, lh = lane >> 4;
  const int wm = wv & 3, wn = wv >> 2;
  const int m0 = blockIdx.y * 128;
  const int n0 = blockIdx.x * 64;

  const int ar_row = tid >> 1, ar_col = (tid & 1) * 16;  // W-tile fill (128 x 32)
  const int bk = tid >> 3, bn = (tid & 7) * 8;           // G-tile fill (32k x 64n), transposed store

  f32x8 cre[2][2], cxx[2][2], cim[2][2];   // ar*br, ai*bi, (ar*bi + ai*br)
  #pragma unroll
  for (int a = 0; a < 2; ++a)
    #pragma unroll
    for (int b = 0; b < 2; ++b)
      #pragma unroll
      for (int v = 0; v < 8; ++v) { cre[a][b][v] = 0.f; cxx[a][b][v] = 0.f; cim[a][b][v] = 0.f; }

  bf16x16 rWr, rWi;
  float4 g0r, g1r, g0i, g1i;
  {
    rWr = *(const bf16x16*)(Wr + (size_t)(m0 + ar_row) * ARRAY + ar_col);
    rWi = *(const bf16x16*)(Wi + (size_t)(m0 + ar_row) * ARRAY + ar_col);
    const float* pr = Gr + (size_t)bk * LENS + n0 + bn;
    const float* pi = Gi + (size_t)bk * LENS + n0 + bn;
    g0r = *(const float4*)pr; g1r = *(const float4*)(pr + 4);
    g0i = *(const float4*)pi; g1i = *(const float4*)(pi + 4);
  }

  const int NK = ARRAY / 32;
  for (int kk = 0; kk < NK; ++kk) {
    const int buf = kk & 1;
    st_row32(&sW[buf][0][ar_row * LDA + ar_col], rWr);
    st_row32(&sW[buf][1][ar_row * LDA + ar_col], rWi);
    {
      const float fr[8] = {g0r.x, g0r.y, g0r.z, g0r.w, g1r.x, g1r.y, g1r.z, g1r.w};
      const float fi[8] = {g0i.x, g0i.y, g0i.z, g0i.w, g1i.x, g1i.y, g1i.z, g1i.w};
      #pragma unroll
      for (int j = 0; j < 8; ++j) {
        sB[buf][0][(bn + j) * LDA + bk] = (__bf16)fr[j];
        sB[buf][1][(bn + j) * LDA + bk] = (__bf16)fi[j];
      }
    }
    __syncthreads();
    if (kk + 1 < NK) {  // prefetch next chunk; wait lands after the WMMA block
      const int kb = (kk + 1) * 32;
      rWr = *(const bf16x16*)(Wr + (size_t)(m0 + ar_row) * ARRAY + kb + ar_col);
      rWi = *(const bf16x16*)(Wi + (size_t)(m0 + ar_row) * ARRAY + kb + ar_col);
      const float* pr = Gr + (size_t)(kb + bk) * LENS + n0 + bn;
      const float* pi = Gi + (size_t)(kb + bk) * LENS + n0 + bn;
      g0r = *(const float4*)pr; g1r = *(const float4*)(pr + 4);
      g0i = *(const float4*)pi; g1i = *(const float4*)(pi + 4);
    }
    bf16x16 arf[2], aif[2], brf[2], bif[2];
    #pragma unroll
    for (int mi = 0; mi < 2; ++mi) {
      const int row = wm * 32 + mi * 16 + ln;
      arf[mi] = ld_frag_a(sW[buf][0], row, lh);
      aif[mi] = ld_frag_a(sW[buf][1], row, lh);
    }
    #pragma unroll
    for (int ni = 0; ni < 2; ++ni) {
      const int row = wn * 32 + ni * 16 + ln;
      brf[ni] = ld_frag_b(sB[buf][0], row, lh);
      bif[ni] = ld_frag_b(sB[buf][1], row, lh);
    }
    #pragma unroll
    for (int mi = 0; mi < 2; ++mi)
      #pragma unroll
      for (int ni = 0; ni < 2; ++ni) {
        cre[mi][ni] = wmma_acc(arf[mi], brf[ni], cre[mi][ni]);
        cxx[mi][ni] = wmma_acc(aif[mi], bif[ni], cxx[mi][ni]);
        cim[mi][ni] = wmma_acc(arf[mi], bif[ni], cim[mi][ni]);
        cim[mi][ni] = wmma_acc(aif[mi], brf[ni], cim[mi][ni]);
      }
  }

  // epilogue: re = cre - cxx, theta column scale (C layout: n = lane&15, m = v + 8*(lane>>4))
  #pragma unroll
  for (int ni = 0; ni < 2; ++ni) {
    const int n = n0 + wn * 32 + ni * 16 + ln;
    const float2 th = Th[n];
    #pragma unroll
    for (int mi = 0; mi < 2; ++mi) {
      #pragma unroll
      for (int v = 0; v < 8; ++v) {
        const int m = m0 + wm * 32 + mi * 16 + v + 8 * lh;
        const float re = cre[mi][ni][v] - cxx[mi][ni][v];
        const float im = cim[mi][ni][v];
        const size_t off = (size_t)m * LENS + n;
        Lr[off] = (__bf16)(re * th.x - im * th.y);
        Li[off] = (__bf16)(re * th.y + im * th.x);
      }
    }
  }
}

// ---------------- GEMM2: sout = lens @ A^T, fused |.|^2 / diagonal reductions ----------------
__global__ __launch_bounds__(256) void k_gemm2(
    const __bf16* __restrict__ Lr, const __bf16* __restrict__ Li,
    const float*  __restrict__ Ar, const float*  __restrict__ Ai,
    float* __restrict__ dRe, float* __restrict__ dIm,
    float* __restrict__ dP1, float* __restrict__ dM1,
    float* __restrict__ partial) {
  __shared__ __bf16 sL[2][2][128 * LDA];
  __shared__ __bf16 sA[2][2][64 * LDA];
  __shared__ float sRed[8];

  const int tid = threadIdx.x;
  const int lane = tid & 31, wv = tid >> 5;
  const int ln = lane & 15, lh = lane >> 4;
  const int wm = wv & 3, wn = wv >> 2;
  const int m0 = blockIdx.y * 128;
  const int n0 = blockIdx.x * 64;

  const int lr_row = tid >> 1, lr_col = (tid & 1) * 16;  // lens fill (128 x 32)
  const int an = tid >> 2, ak = (tid & 3) * 8;           // A fill (64n x 32k), no transpose needed

  f32x8 cre[2][2], cxx[2][2], cim[2][2];
  #pragma unroll
  for (int a = 0; a < 2; ++a)
    #pragma unroll
    for (int b = 0; b < 2; ++b)
      #pragma unroll
      for (int v = 0; v < 8; ++v) { cre[a][b][v] = 0.f; cxx[a][b][v] = 0.f; cim[a][b][v] = 0.f; }

  bf16x16 rLr, rLi;
  float4 a0r, a1r, a0i, a1i;
  {
    rLr = *(const bf16x16*)(Lr + (size_t)(m0 + lr_row) * LENS + lr_col);
    rLi = *(const bf16x16*)(Li + (size_t)(m0 + lr_row) * LENS + lr_col);
    const float* pr = Ar + (size_t)(n0 + an) * LENS + ak;
    const float* pi = Ai + (size_t)(n0 + an) * LENS + ak;
    a0r = *(const float4*)pr; a1r = *(const float4*)(pr + 4);
    a0i = *(const float4*)pi; a1i = *(const float4*)(pi + 4);
  }

  const int NK = LENS / 32;
  for (int kk = 0; kk < NK; ++kk) {
    const int buf = kk & 1;
    st_row32(&sL[buf][0][lr_row * LDA + lr_col], rLr);
    st_row32(&sL[buf][1][lr_row * LDA + lr_col], rLi);
    {
      const float fr[8] = {a0r.x, a0r.y, a0r.z, a0r.w, a1r.x, a1r.y, a1r.z, a1r.w};
      const float fi[8] = {a0i.x, a0i.y, a0i.z, a0i.w, a1i.x, a1i.y, a1i.z, a1i.w};
      bf16x8 hr, hi_;
      #pragma unroll
      for (int j = 0; j < 8; ++j) { hr[j] = (__bf16)fr[j]; hi_[j] = (__bf16)fi[j]; }
      *(bf16x8*)(&sA[buf][0][an * LDA + ak]) = hr;
      *(bf16x8*)(&sA[buf][1][an * LDA + ak]) = hi_;
    }
    __syncthreads();
    if (kk + 1 < NK) {
      const int kb = (kk + 1) * 32;
      rLr = *(const bf16x16*)(Lr + (size_t)(m0 + lr_row) * LENS + kb + lr_col);
      rLi = *(const bf16x16*)(Li + (size_t)(m0 + lr_row) * LENS + kb + lr_col);
      const float* pr = Ar + (size_t)(n0 + an) * LENS + kb + ak;
      const float* pi = Ai + (size_t)(n0 + an) * LENS + kb + ak;
      a0r = *(const float4*)pr; a1r = *(const float4*)(pr + 4);
      a0i = *(const float4*)pi; a1i = *(const float4*)(pi + 4);
    }
    bf16x16 arf[2], aif[2], brf[2], bif[2];
    #pragma unroll
    for (int mi = 0; mi < 2; ++mi) {
      const int row = wm * 32 + mi * 16 + ln;
      arf[mi] = ld_frag_a(sL[buf][0], row, lh);
      aif[mi] = ld_frag_a(sL[buf][1], row, lh);
    }
    #pragma unroll
    for (int ni = 0; ni < 2; ++ni) {
      const int row = wn * 32 + ni * 16 + ln;
      brf[ni] = ld_frag_b(sA[buf][0], row, lh);
      bif[ni] = ld_frag_b(sA[buf][1], row, lh);
    }
    #pragma unroll
    for (int mi = 0; mi < 2; ++mi)
      #pragma unroll
      for (int ni = 0; ni < 2; ++ni) {
        cre[mi][ni] = wmma_acc(arf[mi], brf[ni], cre[mi][ni]);
        cxx[mi][ni] = wmma_acc(aif[mi], bif[ni], cxx[mi][ni]);
        cim[mi][ni] = wmma_acc(arf[mi], bif[ni], cim[mi][ni]);
        cim[mi][ni] = wmma_acc(aif[mi], brf[ni], cim[mi][ni]);
      }
  }

  // fused epilogue: |sout|^2 totals + three diagonals, no materialization of sout
  float lsum = 0.f;
  #pragma unroll
  for (int ni = 0; ni < 2; ++ni) {
    const int gn = n0 + wn * 32 + ni * 16 + ln;
    #pragma unroll
    for (int mi = 0; mi < 2; ++mi) {
      #pragma unroll
      for (int v = 0; v < 8; ++v) {
        const int gm = m0 + wm * 32 + mi * 16 + v + 8 * lh;
        const float re = cre[mi][ni][v] - cxx[mi][ni][v];
        const float im = cim[mi][ni][v];
        const float a2 = re * re + im * im;
        lsum += a2;
        if (gm == gn)     { dRe[gm] = re; dIm[gm] = im; }
        if (gn == gm + 1) { dP1[gm] = a2; }
        if (gm == gn + 1) { dM1[gn] = a2; }
      }
    }
  }
  #pragma unroll
  for (int o = 16; o > 0; o >>= 1) lsum += __shfl_xor(lsum, o, 32);
  if (lane == 0) sRed[wv] = lsum;
  __syncthreads();
  if (tid == 0) {
    float s = 0.f;
    #pragma unroll
    for (int i = 0; i < 8; ++i) s += sRed[i];
    partial[blockIdx.y * gridDim.x + blockIdx.x] = s;
  }
}

// ---------------- finalize: deterministic reductions -> scalar ----------------
__global__ __launch_bounds__(256) void k_finalize(
    const float* __restrict__ dRe, const float* __restrict__ dIm,
    const float* __restrict__ dP1, const float* __restrict__ dM1,
    const float* __restrict__ partial, int nPart,
    const float* __restrict__ awr, const float* __restrict__ awi,
    float* __restrict__ out) {
  __shared__ float red[256];
  const int tid = threadIdx.x;

  float tot = 0.f;
  for (int i = tid; i < nPart; i += 256) tot += partial[i];

  float d0 = 0.f, svr = 0.f, svi = 0.f, sv2 = 0.f;
  for (int i = tid; i < CODE; i += 256) {
    const float re = dRe[i], im = dIm[i];
    const float a2 = re * re + im * im;
    d0 += a2;
    const float mag = sqrtf(a2);
    const float vr = mag * awr[i], vi = mag * awi[i];
    svr += vr; svi += vi; sv2 += vr * vr + vi * vi;
  }
  float dp = 0.f, dm = 0.f;
  for (int i = tid; i < CODE - 1; i += 256) { dp += dP1[i]; dm += dM1[i]; }

  auto bsum = [&](float v) -> float {
    red[tid] = v; __syncthreads();
    for (int s = 128; s > 0; s >>= 1) {
      if (tid < s) red[tid] += red[tid + s];
      __syncthreads();
    }
    const float r = red[0];
    __syncthreads();
    return r;
  };
  tot = bsum(tot); d0 = bsum(d0); dp = bsum(dp); dm = bsum(dm);
  svr = bsum(svr); svi = bsum(svi); sv2 = bsum(sv2);

  if (tid == 0) {
    const float K = 0.1f;
    const float diag_out = d0 + 0.8f * (dp + dm);
    const float diag_sum = d0 + dp + dm;
    const float n = (float)CODE;
    const float var = (sv2 - (svr * svr + svi * svi) / n) / (n - 1.0f);
    out[0] = -(1.0f + K) * diag_out + K * (tot - diag_sum) + 1000.0f * var;
  }
}

extern "C" void kernel_launch(void* const* d_in, const int* in_sizes, int n_in,
                              void* d_out, int out_size, void* d_ws, size_t ws_size,
                              hipStream_t stream) {
  (void)in_sizes; (void)n_in; (void)out_size; (void)ws_size;

  const float* w_re  = (const float*)d_in[0];
  const float* w_im  = (const float*)d_in[1];
  const float* th_re = (const float*)d_in[2];
  const float* th_im = (const float*)d_in[3];
  const float* A_re  = (const float*)d_in[4];
  const float* A_im  = (const float*)d_in[5];
  const float* G_re  = (const float*)d_in[6];
  const float* G_im  = (const float*)d_in[7];
  const float* aw_re = (const float*)d_in[8];
  const float* aw_im = (const float*)d_in[9];

  char* p = (char*)d_ws;
  __bf16* wnr = (__bf16*)p; p += (size_t)CODE * ARRAY * sizeof(__bf16);
  __bf16* wni = (__bf16*)p; p += (size_t)CODE * ARRAY * sizeof(__bf16);
  float2* thn = (float2*)p; p += (size_t)LENS * sizeof(float2);
  __bf16* lr  = (__bf16*)p; p += (size_t)CODE * LENS * sizeof(__bf16);
  __bf16* li  = (__bf16*)p; p += (size_t)CODE * LENS * sizeof(__bf16);
  float* dRe  = (float*)p; p += (size_t)CODE * sizeof(float);
  float* dIm  = (float*)p; p += (size_t)CODE * sizeof(float);
  float* dP1  = (float*)p; p += (size_t)CODE * sizeof(float);
  float* dM1  = (float*)p; p += (size_t)CODE * sizeof(float);
  float* partial = (float*)p;

  k_prep_w<<<(CODE * ARRAY) / 256, 256, 0, stream>>>(w_re, w_im, wnr, wni);
  k_prep_theta<<<LENS / 256, 256, 0, stream>>>(th_re, th_im, thn);

  dim3 g1(LENS / 64, CODE / 128);   // 128 x 16 blocks
  k_gemm1<<<g1, 256, 0, stream>>>(wnr, wni, G_re, G_im, thn, lr, li);

  dim3 g2(CODE / 64, CODE / 128);   // 32 x 16 blocks
  k_gemm2<<<g2, 256, 0, stream>>>(lr, li, A_re, A_im, dRe, dIm, dP1, dM1, partial);

  const int nPart = (CODE / 64) * (CODE / 128);
  k_finalize<<<1, 256, 0, stream>>>(dRe, dIm, dP1, dM1, partial, nPart, aw_re, aw_im,
                                    (float*)d_out);
}
